// MultiHeadAttention_61357902791348
// MI455X (gfx1250) — matrix operand order repaired and checked by reading, not compiled, not run on hardware
//
#include <hip/hip_runtime.h>
#include <hip/hip_bf16.h>
#include <math.h>

typedef __bf16 bf16;
typedef __bf16 bf16x8  __attribute__((ext_vector_type(8)));
typedef __bf16 bf16x16 __attribute__((ext_vector_type(16)));
typedef float  f32x8   __attribute__((ext_vector_type(8)));

#define N_TOK  3200
#define DMODEL 1536
#define NHEADS 12
#define HDIM   128

// ---------------------------------------------------------------------------
// WMMA helpers (CDNA5 wave32, D = A(16x32 bf16) * B(32x16 bf16) + C(16x16 f32))
// ---------------------------------------------------------------------------
__device__ __forceinline__ f32x8 wmma_bf16f32(bf16x16 a, bf16x16 b, f32x8 c) {
  return __builtin_amdgcn_wmma_f32_16x16x32_bf16(false, a, false, b, (short)0, c,
                                                 false, false);
}

// A-fragment: per-lane row `row` (caller resolves lane%16), k-window base kbase.
// 16-bit A layout: e<8 -> k = kbase + 8*half + e ; e>=8 -> k = kbase + 16 + 8*half + (e-8)
__device__ __forceinline__ bf16x16 load_frag_a(const bf16* p, int stride, int row,
                                               int kbase, int hf) {
  union { bf16x16 v; bf16x8 h[2]; } u;
  const bf16* q = p + row * stride + kbase + hf * 8;
  u.h[0] = *(const bf16x8*)(q);
  u.h[1] = *(const bf16x8*)(q + 16);
  return u.v;
}

// B-fragment from N-major (transposed) storage: lane holds column `col`,
// e<8 -> k = kbase + 8*half + e ; e>=8 -> k = kbase + 16 + 8*half + (e-8)
__device__ __forceinline__ bf16x16 load_frag_b(const bf16* p, int stride, int col,
                                               int kbase, int hf) {
  union { bf16x16 v; bf16x8 h[2]; } u;
  const bf16* q = p + col * stride + kbase + hf * 8;
  u.h[0] = *(const bf16x8*)(q);
  u.h[1] = *(const bf16x8*)(q + 16);
  return u.v;
}

// ---------------------------------------------------------------------------
// Pack / transpose / freqs-table kernels
// ---------------------------------------------------------------------------
__global__ void cast_to_bf16_kernel(const float* __restrict__ in,
                                    bf16* __restrict__ out, int n) {
  int i = blockIdx.x * blockDim.x + threadIdx.x;
  if (i < n) out[i] = (bf16)in[i];
}

// out[n][k] = (bf16) in[k][n]   (dim x dim, dim % 32 == 0), 256 threads
__global__ void transpose_to_bf16_kernel(const float* __restrict__ in,
                                         bf16* __restrict__ out, int dim) {
  __shared__ float tile[32][33];
  int bx = blockIdx.x * 32, by = blockIdx.y * 32;
  int tx = threadIdx.x & 31, ty = threadIdx.x >> 5;   // ty in 0..7
  for (int r = ty; r < 32; r += 8)
    tile[r][tx] = in[(size_t)(by + r) * dim + bx + tx];
  __syncthreads();
  for (int r = ty; r < 32; r += 8)
    out[(size_t)(bx + r) * dim + by + tx] = (bf16)tile[tx][r];
}

// fused (cos,sin) table per token: [N][64][2]; 22 t-dims, 21 h-dims, 21 w-dims
__global__ void build_freqs_kernel(const float* __restrict__ ft,
                                   const float* __restrict__ fh,
                                   const float* __restrict__ fw,
                                   float* __restrict__ out) {
  int n = blockIdx.x;
  int i = threadIdx.x;   // 0..63
  int f = n / 400, rem = n % 400, h = rem / 20, w = rem % 20;
  float c, s;
  if (i < 22)      { c = ft[(f * 22 + i) * 2];        s = ft[(f * 22 + i) * 2 + 1]; }
  else if (i < 43) { c = fh[(h * 21 + (i - 22)) * 2]; s = fh[(h * 21 + (i - 22)) * 2 + 1]; }
  else             { c = fw[(w * 21 + (i - 43)) * 2]; s = fw[(w * 21 + (i - 43)) * 2 + 1]; }
  out[(n * 64 + i) * 2]     = c;
  out[(n * 64 + i) * 2 + 1] = s;
}

// ---------------------------------------------------------------------------
// GEMM: C[M x Nc] = A[M x K](bf16) * Bt[Nc x K]^T(bf16) + bias; 128x128 tile,
// 8 waves (2 x 4), K-step 32, register-prefetch pipeline.
// ---------------------------------------------------------------------------
template <bool OUT_F32>
__launch_bounds__(256)
__global__ void gemm_bf16_wmma(const bf16* __restrict__ A, const bf16* __restrict__ Bt,
                               const float* __restrict__ bias,
                               bf16* __restrict__ Ob, float* __restrict__ Of,
                               int M, int Nc, int K) {
  __shared__ bf16 As[128 * 32];
  __shared__ bf16 Bs[128 * 32];
  const int tid = threadIdx.x;
  const int lane = tid & 31, wave = tid >> 5;
  const int hf = lane >> 4, l16 = lane & 15;
  const int wm = wave & 1;    // 64-row stripe
  const int wn = wave >> 1;   // 32-col stripe
  const int mb = blockIdx.y * 128;
  const int nb = blockIdx.x * 128;

  f32x8 acc[4][2] = {};
  const int nkt = K >> 5;

  // per-thread global->LDS chunks: idx in [0,512): row = idx>>2, off = (idx&3)*8
  bf16x8 ra[2], rb[2];
  for (int i = 0; i < 2; ++i) {
    int idx = i * 256 + tid, row = idx >> 2, off = (idx & 3) * 8;
    ra[i] = *(const bf16x8*)(A  + (size_t)(mb + row) * K + off);
    rb[i] = *(const bf16x8*)(Bt + (size_t)(nb + row) * K + off);
  }

  for (int kt = 0; kt < nkt; ++kt) {
    for (int i = 0; i < 2; ++i) {
      int idx = i * 256 + tid, row = idx >> 2, off = (idx & 3) * 8;
      *(bf16x8*)(&As[row * 32 + off]) = ra[i];
      *(bf16x8*)(&Bs[row * 32 + off]) = rb[i];
    }
    __syncthreads();
    if (kt + 1 < nkt) {
      for (int i = 0; i < 2; ++i) {
        int idx = i * 256 + tid, row = idx >> 2, off = (idx & 3) * 8;
        ra[i] = *(const bf16x8*)(A  + (size_t)(mb + row) * K + (kt + 1) * 32 + off);
        rb[i] = *(const bf16x8*)(Bt + (size_t)(nb + row) * K + (kt + 1) * 32 + off);
      }
    }
    bf16x16 af[4], bfr[2];
    for (int i = 0; i < 4; ++i) af[i]  = load_frag_a(As, 32, wm * 64 + i * 16 + l16, 0, hf);
    for (int j = 0; j < 2; ++j) bfr[j] = load_frag_b(Bs, 32, wn * 32 + j * 16 + l16, 0, hf);
    for (int i = 0; i < 4; ++i)
      for (int j = 0; j < 2; ++j)
        acc[i][j] = wmma_bf16f32(af[i], bfr[j], acc[i][j]);
    __syncthreads();
  }

  for (int i = 0; i < 4; ++i)
    for (int j = 0; j < 2; ++j) {
      int col = nb + wn * 32 + j * 16 + l16;
      float bv = bias[col];
      for (int e = 0; e < 8; ++e) {
        int row = mb + wm * 64 + i * 16 + e + hf * 8;   // C layout: M = e + 8*half
        float v = acc[i][j][e] + bv;
        if constexpr (OUT_F32) Of[(size_t)row * Nc + col] = v;
        else                   Ob[(size_t)row * Nc + col] = (bf16)v;
      }
    }
}

// ---------------------------------------------------------------------------
// Fused RMSNorm (over HD=128) + 3-axis RoPE, in place on bf16 [N][NH*HD].
// One wave per (token, head); 8 waves per block.
// ---------------------------------------------------------------------------
__launch_bounds__(256)
__global__ void rmsnorm_rope_kernel(bf16* __restrict__ qk, const float* __restrict__ scale,
                                    const float* __restrict__ freqs) {
  int gw = blockIdx.x * 8 + (threadIdx.x >> 5);
  if (gw >= N_TOK * NHEADS) return;
  int n = gw / NHEADS, h = gw % NHEADS;
  int lane = threadIdx.x & 31;
  bf16* base = qk + (size_t)n * DMODEL + h * HDIM;

  float x0[2], x1[2], ss = 0.f;
  for (int t = 0; t < 2; ++t) {
    int p = lane * 2 + t;              // pair index 0..63
    x0[t] = (float)base[2 * p];
    x1[t] = (float)base[2 * p + 1];
    ss += x0[t] * x0[t] + x1[t] * x1[t];
  }
  for (int m = 16; m >= 1; m >>= 1) ss += __shfl_xor(ss, m, 32);
  float r = rsqrtf(ss * (1.0f / 128.0f) + 1e-6f);
  for (int t = 0; t < 2; ++t) {
    int p = lane * 2 + t;
    float a = x0[t] * r * scale[2 * p];
    float b = x1[t] * r * scale[2 * p + 1];
    float c = freqs[(n * 64 + p) * 2];
    float s = freqs[(n * 64 + p) * 2 + 1];
    base[2 * p]     = (bf16)(a * c - b * s);
    base[2 * p + 1] = (bf16)(a * s + b * c);
  }
}

// ---------------------------------------------------------------------------
// Flash attention: block = (head, 128 q-rows), 8 waves x 16 rows each.
// K tile 64x128 and V^T tile 128x64 staged in LDS; online softmax in f32.
// ---------------------------------------------------------------------------
__launch_bounds__(256)
__global__ void flash_attn_wmma(const bf16* __restrict__ Q, const bf16* __restrict__ Km,
                                const bf16* __restrict__ Vm, bf16* __restrict__ O) {
  constexpr int LDK = 136;   // K tile row stride (d=128 + pad 8, keeps 16B align)
  constexpr int LDV = 72;    // V^T / P row stride (key=64 + pad 8)
  constexpr float SCALE = 0.08838834764831845f;   // 1/sqrt(128)
  __shared__ bf16 Ks[64 * LDK];
  __shared__ bf16 Vts[128 * LDV];
  __shared__ bf16 Ps[8 * 16 * LDV];

  const int tid = threadIdx.x;
  const int lane = tid & 31, wave = tid >> 5;
  const int hf = lane >> 4, l16 = lane & 15;
  const int h = blockIdx.x;
  const int q0 = blockIdx.y * 128;

  // Q fragments for this wave's 16 rows, loaded from global once per block.
  bf16x16 qf[4];
  {
    const bf16* qrow = Q + (size_t)(q0 + wave * 16 + l16) * DMODEL + h * HDIM;
    for (int kk = 0; kk < 4; ++kk) {
      union { bf16x16 v; bf16x8 hh[2]; } u;
      const bf16* p = qrow + kk * 32 + hf * 8;
      u.hh[0] = *(const bf16x8*)(p);
      u.hh[1] = *(const bf16x8*)(p + 16);
      qf[kk] = u.v;
    }
  }

  float mrow[8], lrow[8];
  f32x8 of[8] = {};
  for (int e = 0; e < 8; ++e) { mrow[e] = -__builtin_inff(); lrow[e] = 0.f; }
  bf16* Pw = Ps + wave * 16 * LDV;

  for (int kt = 0; kt < N_TOK / 64; ++kt) {
    const int k0 = kt * 64;
    // Stage K tile [64 key][128 d] and transposed V tile [128 d][64 key].
    for (int it = 0; it < 4; ++it) {
      int idx = it * 256 + tid;          // 1024 chunks of 16B
      int key = idx >> 4, ch = idx & 15;
      bf16x8 kv = *(const bf16x8*)(Km + (size_t)(k0 + key) * DMODEL + h * HDIM + ch * 8);
      *(bf16x8*)(&Ks[key * LDK + ch * 8]) = kv;
      bf16x8 vv = *(const bf16x8*)(Vm + (size_t)(k0 + key) * DMODEL + h * HDIM + ch * 8);
      for (int j = 0; j < 8; ++j) Vts[(ch * 8 + j) * LDV + key] = vv[j];
    }
    if (kt + 1 < N_TOK / 64)   // hint next K tile into cache (global_prefetch)
      __builtin_prefetch(Km + (size_t)(k0 + 64 + (tid >> 2)) * DMODEL + h * HDIM, 0, 1);
    __syncthreads();

    // S = Q K^T (16 rows x 64 keys per wave), f32 accum, pre-scaled.
    f32x8 sf[4];
    for (int nf = 0; nf < 4; ++nf) {
      f32x8 acc = {};
      for (int kk = 0; kk < 4; ++kk) {
        bf16x16 kb = load_frag_b(Ks, LDK, nf * 16 + l16, kk * 32, hf);
        acc = wmma_bf16f32(qf[kk], kb, acc);
      }
      for (int e = 0; e < 8; ++e) acc[e] *= SCALE;
      sf[nf] = acc;
    }

    // Online softmax per row (row m = e + 8*half; replicated across 16 lanes of half).
    for (int e = 0; e < 8; ++e) {
      float tm = sf[0][e];
      for (int nf = 1; nf < 4; ++nf) tm = fmaxf(tm, sf[nf][e]);
      for (int m = 1; m < 16; m <<= 1) tm = fmaxf(tm, __shfl_xor(tm, m, 32));
      float mn = fmaxf(mrow[e], tm);
      float corr = __expf(mrow[e] - mn);
      mrow[e] = mn;
      float rs = 0.f;
      for (int nf = 0; nf < 4; ++nf) {
        float p = __expf(sf[nf][e] - mn);
        rs += p;
        Pw[(e + hf * 8) * LDV + nf * 16 + l16] = (bf16)p;   // C-layout -> LDS
      }
      for (int m = 1; m < 16; m <<= 1) rs += __shfl_xor(rs, m, 32);
      lrow[e] = lrow[e] * corr + rs;
      for (int df = 0; df < 8; ++df) of[df][e] *= corr;
    }

    // O += P * V  (P 16x64 as A-frags from LDS; V^T rows are d -> B-frags).
    for (int df = 0; df < 8; ++df)
      for (int kk = 0; kk < 2; ++kk) {
        bf16x16 pa = load_frag_a(Pw, LDV, l16, kk * 32, hf);
        bf16x16 vb = load_frag_b(Vts, LDV, df * 16 + l16, kk * 32, hf);
        of[df] = wmma_bf16f32(pa, vb, of[df]);
      }
    __syncthreads();
  }

  for (int df = 0; df < 8; ++df)
    for (int e = 0; e < 8; ++e) {
      int row = q0 + wave * 16 + e + hf * 8;
      int col = h * HDIM + df * 16 + l16;
      O[(size_t)row * DMODEL + col] = (bf16)(of[df][e] / lrow[e]);
    }
}

// ---------------------------------------------------------------------------
// Launch
// ---------------------------------------------------------------------------
extern "C" void kernel_launch(void* const* d_in, const int* in_sizes, int n_in,
                              void* d_out, int out_size, void* d_ws, size_t ws_size,
                              hipStream_t stream) {
  (void)in_sizes; (void)n_in; (void)out_size; (void)ws_size;
  const float* x  = (const float*)d_in[0];
  const float* Wq = (const float*)d_in[1];
  const float* bq = (const float*)d_in[2];
  const float* Wk = (const float*)d_in[3];
  const float* bk = (const float*)d_in[4];
  const float* Wv = (const float*)d_in[5];
  const float* bv = (const float*)d_in[6];
  const float* Wo = (const float*)d_in[7];
  const float* bo = (const float*)d_in[8];
  const float* qsc = (const float*)d_in[9];
  const float* ksc = (const float*)d_in[10];
  const float* ft = (const float*)d_in[11];
  const float* fh = (const float*)d_in[12];
  const float* fw = (const float*)d_in[13];
  float* out = (float*)d_out;

  size_t off = 0;
  auto alloc = [&](size_t bytes) -> void* {
    void* p = (char*)d_ws + off;
    off += (bytes + 255) & ~(size_t)255;
    return p;
  };
  const size_t ND = (size_t)N_TOK * DMODEL;
  const size_t DD = (size_t)DMODEL * DMODEL;
  bf16*  xb  = (bf16*)alloc(ND * 2);
  bf16*  wqt = (bf16*)alloc(DD * 2);
  bf16*  wkt = (bf16*)alloc(DD * 2);
  bf16*  wvt = (bf16*)alloc(DD * 2);
  bf16*  wot = (bf16*)alloc(DD * 2);
  bf16*  qb  = (bf16*)alloc(ND * 2);
  bf16*  kb  = (bf16*)alloc(ND * 2);
  bf16*  vb  = (bf16*)alloc(ND * 2);
  bf16*  ab  = (bf16*)alloc(ND * 2);
  float* fr  = (float*)alloc((size_t)N_TOK * 64 * 2 * 4);

  cast_to_bf16_kernel<<<(int)((ND + 255) / 256), 256, 0, stream>>>(x, xb, (int)ND);

  dim3 tg(DMODEL / 32, DMODEL / 32);
  transpose_to_bf16_kernel<<<tg, 256, 0, stream>>>(Wq, wqt, DMODEL);
  transpose_to_bf16_kernel<<<tg, 256, 0, stream>>>(Wk, wkt, DMODEL);
  transpose_to_bf16_kernel<<<tg, 256, 0, stream>>>(Wv, wvt, DMODEL);
  transpose_to_bf16_kernel<<<tg, 256, 0, stream>>>(Wo, wot, DMODEL);

  build_freqs_kernel<<<N_TOK, 64, 0, stream>>>(ft, fh, fw, fr);

  dim3 gg(DMODEL / 128, N_TOK / 128);
  gemm_bf16_wmma<false><<<gg, 256, 0, stream>>>(xb, wqt, bq, qb, nullptr, N_TOK, DMODEL, DMODEL);
  gemm_bf16_wmma<false><<<gg, 256, 0, stream>>>(xb, wkt, bk, kb, nullptr, N_TOK, DMODEL, DMODEL);
  gemm_bf16_wmma<false><<<gg, 256, 0, stream>>>(xb, wvt, bv, vb, nullptr, N_TOK, DMODEL, DMODEL);

  rmsnorm_rope_kernel<<<(N_TOK * NHEADS) / 8, 256, 0, stream>>>(qb, qsc, fr);
  rmsnorm_rope_kernel<<<(N_TOK * NHEADS) / 8, 256, 0, stream>>>(kb, ksc, fr);

  flash_attn_wmma<<<dim3(NHEADS, N_TOK / 128), 256, 0, stream>>>(qb, kb, vb, ab);

  gemm_bf16_wmma<true><<<gg, 256, 0, stream>>>(ab, wot, bo, nullptr, out, N_TOK, DMODEL, DMODEL);
}
